// ConceptDiagram_2370821948079
// MI455X (gfx1250) — matrix operand order repaired and checked by reading, not compiled, not run on hardware
//
#include <hip/hip_runtime.h>
#include <math.h>

// ---------------------------------------------------------------------------
// MetaphorMorphism forward, routed-expert version for MI455X (gfx1250).
//   N=65536 items, D=H=DT=128, S=T=4.
// Strategy: bucket items by type_ids (4 experts), run each bucket through
// 4 WMMA layers (mapper D->H, H->DT, classifier D->H, H->1) using
// v_wmma_f32_16x16x32_bf16 (bf16 in / f32 accum): the roofline needs ~2.1
// PFLOP/s of matmul to stay at the 23.3 TB/s HBM roof, which only the bf16
// matrix path provides. Weights are repacked once per launch into the
// per-lane WMMA B-fragment layout so every B fetch is one contiguous
// 32B/lane load from L2 (weights are ~400KB, L2 is 192MB -> always hot).
// The classifier column C2 is packed as a B-fragment with only column n=0
// non-zero, so the 128-long dot product is 4 WMMAs instead of a VALU loop.
// ---------------------------------------------------------------------------

typedef __attribute__((ext_vector_type(16))) __bf16 v16bf;
typedef __attribute__((ext_vector_type(8)))  __bf16 v8bf;
typedef __attribute__((ext_vector_type(4)))  __bf16 v4bf;
typedef __attribute__((ext_vector_type(8)))  float  v8f;

#define S_TYPES   4
#define T_TARGETS 4
#define DIM       128
#define ROWSTRIDE 136   // bf16 units; 272B row stride staggers LDS banks (+4/row)

// ---- workspace layout (bytes); total ~681 KB, assume ws_size >= that ----
enum : size_t {
  WS_BEST_T  = 0,                     // int[4]
  WS_BEST_P  = 16,                    // float[4] sigmoid(best logit)
  WS_COUNTS  = 32,                    // int[4]
  WS_CURSOR  = 48,                    // int[4]
  WS_OFFSETS = 64,                    // int[4]
  WS_B1      = 128,                   // float[4][128]
  WS_B2      = WS_B1  + 2048,         // float[4][128]
  WS_C1BIAS  = WS_B2  + 2048,         // float[4][128]
  WS_C2S     = WS_C1BIAS + 2048,      // float[4] classifier output bias
  WS_W1      = 8704,                  // __bf16[4][16384] (B-frag layout)
  WS_W2      = WS_W1 + 131072,
  WS_C1M     = WS_W2 + 131072,
  WS_IDX     = WS_C1M + 131072,       // int[N] compacted item indices
  WS_C2FRAG  = WS_IDX + 262144,       // __bf16[4][4*512] C2 column as B-frags
};

__device__ __forceinline__ int imin(int a, int b) { return a < b ? a : b; }

// ---------------------------------------------------------------------------
// Kernel 0: argmax routing, bias copies, weight repack f32 -> bf16 B-fragments
// B-frag layout (16x16x32 bf16 WMMA, wave32):
//   lane l holds column n = nb*16 + (l&15); half h holds K = kb*32+(l>>4)*16+h
//   fragment (kb,nb) stored contiguously: 32 lanes x 16 halves = 1 KB
// ---------------------------------------------------------------------------
__global__ void prep_kernel(const float* __restrict__ tm,
                            const float* __restrict__ W1, const float* __restrict__ b1,
                            const float* __restrict__ W2, const float* __restrict__ b2,
                            const float* __restrict__ C1, const float* __restrict__ c1,
                            const float* __restrict__ C2, const float* __restrict__ c2,
                            char* __restrict__ ws)
{
  __shared__ int s_bt[S_TYPES];
  const int tid = threadIdx.x;

  int*    best_t = (int*)  (ws + WS_BEST_T);
  float*  best_p = (float*)(ws + WS_BEST_P);
  int*    counts = (int*)  (ws + WS_COUNTS);
  float*  b1w = (float*)(ws + WS_B1);
  float*  b2w = (float*)(ws + WS_B2);
  float*  c1w = (float*)(ws + WS_C1BIAS);
  float*  c2w = (float*)(ws + WS_C2S);
  __bf16* W1b = (__bf16*)(ws + WS_W1);
  __bf16* W2b = (__bf16*)(ws + WS_W2);
  __bf16* C1b = (__bf16*)(ws + WS_C1M);
  __bf16* C2f = (__bf16*)(ws + WS_C2FRAG);

  if (tid < S_TYPES) {
    const int s = tid;
    int bt = 0; float bv = tm[s * T_TARGETS];
    for (int t = 1; t < T_TARGETS; ++t) {
      float v = tm[s * T_TARGETS + t];
      if (v > bv) { bv = v; bt = t; }
    }
    s_bt[s] = bt;
    best_t[s] = bt;
    best_p[s] = 1.0f / (1.0f + __expf(-bv));
    c2w[s]    = c2[s * T_TARGETS + bt];
  }
  if (tid >= S_TYPES && tid < 2 * S_TYPES) counts[tid - S_TYPES] = 0;
  __syncthreads();

  // biases b1/b2/c1 for the chosen t*
  for (int e = tid; e < 3 * S_TYPES * DIM; e += blockDim.x) {
    const int which = e / (S_TYPES * DIM);
    const int rem   = e % (S_TYPES * DIM);
    const int s = rem / DIM, h = rem % DIM;
    const int src = (s * T_TARGETS + s_bt[s]) * DIM + h;
    if (which == 0)      b1w[rem] = b1[src];
    else if (which == 1) b2w[rem] = b2[src];
    else                 c1w[rem] = c1[src];
  }

  // C2 column packed as B-fragments with only column n==0 carrying data:
  // lane l -> n = l&15; K = kb*32 + (l>>4)*16 + h
  for (int u = tid; u < S_TYPES * 4 * 512; u += blockDim.x) {
    const int s  = u / 2048;
    const int j  = u % 2048;
    const int kb = j / 512;
    const int e  = j % 512;
    const int lane = e / 16, h = e % 16;
    const int n = lane & 15;
    const int k = kb * 32 + (lane >> 4) * 16 + h;
    float val = 0.0f;
    if (n == 0) val = C2[(s * T_TARGETS + s_bt[s]) * DIM + k];  // C2[s,t,k,0]
    C2f[u] = (__bf16)val;
  }

  // weight repack (4 experts x 16384 elements x 3 matrices)
  for (int u = tid; u < S_TYPES * DIM * DIM; u += blockDim.x) {
    const int s = u >> 14;
    const int i = u & (DIM * DIM - 1);
    const int k = i >> 7, n = i & 127;
    const int src = ((s * T_TARGETS + s_bt[s]) * DIM + k) * DIM + n;
    const int kb = k >> 5, r = k & 31, hi = r >> 4, h = r & 15;
    const int lane = (n & 15) | (hi << 4);
    const int nb = n >> 4;
    const int dst = (((kb * 8 + nb) * 32 + lane) * 16 + h) + s * DIM * DIM;
    W1b[dst] = (__bf16)W1[src];
    W2b[dst] = (__bf16)W2[src];
    C1b[dst] = (__bf16)C1[src];
  }
}

// ---------------------------------------------------------------------------
// Bucketing: count -> exclusive scan -> scatter of item indices.
// Intra-bucket order is atomics-nondeterministic, but each item's result is
// independent of its slot, so d_out is deterministic.
// ---------------------------------------------------------------------------
__global__ void count_kernel(const int* __restrict__ type_ids, char* ws, int N) {
  const int i = blockIdx.x * blockDim.x + threadIdx.x;
  if (i < N) atomicAdd((int*)(ws + WS_COUNTS) + type_ids[i], 1);
}

__global__ void scan_kernel(char* ws) {
  const int* counts  = (const int*)(ws + WS_COUNTS);
  int*       offsets = (int*)(ws + WS_OFFSETS);
  int*       cursor  = (int*)(ws + WS_CURSOR);
  int off = 0;
  for (int s = 0; s < S_TYPES; ++s) { offsets[s] = off; cursor[s] = off; off += counts[s]; }
}

__global__ void scatter_kernel(const int* __restrict__ type_ids, char* ws, int N) {
  const int i = blockIdx.x * blockDim.x + threadIdx.x;
  if (i < N) {
    const int s = type_ids[i];
    const int p = atomicAdd((int*)(ws + WS_CURSOR) + s, 1);
    ((int*)(ws + WS_IDX))[p] = i;
  }
}

// ---------------------------------------------------------------------------
// Main fused kernel: 128 threads = 4 wave32, each wave owns an independent
// 16-row tile of one expert bucket. Per wave: 3*32 + 4 = 100 bf16 WMMAs.
// No block barriers: every LDS slab is wave-private (in-order DScnt suffices).
// ---------------------------------------------------------------------------
__global__ void __launch_bounds__(128) expert_kernel(
    const float* __restrict__ states, const float* __restrict__ scores,
    const char*  __restrict__ ws,
    float* __restrict__ out_state, float* __restrict__ out_score,
    float* __restrict__ out_prob, int N, int blocksPerType)
{
  const int*   counts  = (const int*)(ws + WS_COUNTS);
  const int*   offsets = (const int*)(ws + WS_OFFSETS);
  const int*   idxArr  = (const int*)(ws + WS_IDX);
  const float* best_p  = (const float*)(ws + WS_BEST_P);

  const int s    = blockIdx.x / blocksPerType;
  const int blk  = blockIdx.x % blocksPerType;
  const int wave = threadIdx.x >> 5;
  const int lane = threadIdx.x & 31;

  const int count = counts[s];
  const int rowBase = blk * 64 + wave * 16;     // bucket-row of this wave's tile
  if (rowBase >= count) return;                 // wave-uniform -> EXEC stays all-1s
  const int off = offsets[s];

  __shared__ __align__(16) __bf16 sX[4][16 * ROWSTRIDE];  // item tiles, bf16
  __shared__ __align__(16) __bf16 sH[4][16 * ROWSTRIDE];  // hidden tiles, bf16
  __shared__ int sItems[4][16];

  __bf16* sXw = sX[wave];
  __bf16* sHw = sH[wave];

  const __bf16* W1p = (const __bf16*)(ws + WS_W1)     + s * DIM * DIM;
  const __bf16* W2p = (const __bf16*)(ws + WS_W2)     + s * DIM * DIM;
  const __bf16* C1p = (const __bf16*)(ws + WS_C1M)    + s * DIM * DIM;
  const __bf16* C2p = (const __bf16*)(ws + WS_C2FRAG) + s * 2048;
  const float*  b1p = (const float*)(ws + WS_B1)     + s * DIM;
  const float*  b2p = (const float*)(ws + WS_B2)     + s * DIM;
  const float*  c1p = (const float*)(ws + WS_C1BIAS) + s * DIM;
  const float   c2s = ((const float*)(ws + WS_C2S))[s];

  // warm the weight streams (global_prefetch_b8); weights live in L2 (192MB)
  __builtin_prefetch(W1p + lane * 512, 0, 1);
  __builtin_prefetch(W2p + lane * 512, 0, 1);
  __builtin_prefetch(C1p + lane * 512, 0, 1);

  if (lane < 16)
    sItems[wave][lane] = idxArr[off + imin(rowBase + lane, count - 1)];

  // ---- stage 16 gathered item rows into LDS as bf16 (row stride 136) ----
  #pragma unroll
  for (int it = 0; it < 16; ++it) {
    const int e  = it * 32 + lane;     // 512 float4 vectors total
    const int r  = e >> 5;             // 32 vec4 per row
    const int cv = e & 31;
    const int item = idxArr[off + imin(rowBase + r, count - 1)];
    const float4 v = *(const float4*)(states + (size_t)item * DIM + cv * 4);
    v4bf o; o[0] = (__bf16)v.x; o[1] = (__bf16)v.y; o[2] = (__bf16)v.z; o[3] = (__bf16)v.w;
    *(v4bf*)(sXw + r * ROWSTRIDE + cv * 4) = o;
  }

  // A-fragment loader: lane l (<16) row l, halves {K..K+15};
  // lanes 16-31 same rows, K offset +16 (matches B-frag packing).
  auto load_afrags = [&](const __bf16* slab, v16bf af[4]) {
    const int row  = lane & 15;
    const int koff = (lane < 16) ? 0 : 16;
    #pragma unroll
    for (int kb = 0; kb < 4; ++kb) {
      const v8bf lo = *(const v8bf*)(slab + row * ROWSTRIDE + kb * 32 + koff);
      const v8bf hi = *(const v8bf*)(slab + row * ROWSTRIDE + kb * 32 + 8 + koff);
      v16bf a;
      #pragma unroll
      for (int q = 0; q < 8; ++q) { a[q] = lo[q]; a[q + 8] = hi[q]; }
      af[kb] = a;
    }
  };

  const int colLane = lane & 15;
  const int rb      = (lane < 16) ? 0 : 8;   // C/D row base per ISA layout
  v16bf af[4];

  // =============== layer 1: H = relu(X @ W1 + b1) -> LDS bf16 ===============
  load_afrags(sXw, af);
  #pragma unroll
  for (int nb = 0; nb < 8; ++nb) {
    const float bv = b1p[nb * 16 + colLane];
    v8f acc = {bv, bv, bv, bv, bv, bv, bv, bv};
    #pragma unroll
    for (int kb = 0; kb < 4; ++kb) {
      const v16bf bf = *(const v16bf*)(W1p + ((kb * 8 + nb) * 32 + lane) * 16);
      acc = __builtin_amdgcn_wmma_f32_16x16x32_bf16(false, af[kb], false, bf,
                                                    (short)0, acc, false, false);
    }
    const int colb = nb * 16 + colLane;
    #pragma unroll
    for (int v = 0; v < 8; ++v)
      sHw[(rb + v) * ROWSTRIDE + colb] = (__bf16)fmaxf(acc[v], 0.0f);
  }

  // =============== layer 2: Y = H @ W2 + b2 -> out_state (f32) ==============
  load_afrags(sHw, af);
  #pragma unroll
  for (int nb = 0; nb < 8; ++nb) {
    const float bv = b2p[nb * 16 + colLane];
    v8f acc = {bv, bv, bv, bv, bv, bv, bv, bv};
    #pragma unroll
    for (int kb = 0; kb < 4; ++kb) {
      const v16bf bf = *(const v16bf*)(W2p + ((kb * 8 + nb) * 32 + lane) * 16);
      acc = __builtin_amdgcn_wmma_f32_16x16x32_bf16(false, af[kb], false, bf,
                                                    (short)0, acc, false, false);
    }
    const int colb = nb * 16 + colLane;
    #pragma unroll
    for (int v = 0; v < 8; ++v) {
      const int tr = rb + v;
      const int gr = rowBase + tr;
      if (gr < count) {
        const int item = sItems[wave][tr];
        out_state[(size_t)item * DIM + colb] = acc[v];  // half-wave 64B segments
      }
    }
  }

  // =============== layer 3: Hc = relu(X @ C1 + c1) -> LDS bf16 ==============
  load_afrags(sXw, af);
  #pragma unroll
  for (int nb = 0; nb < 8; ++nb) {
    const float bv = c1p[nb * 16 + colLane];
    v8f acc = {bv, bv, bv, bv, bv, bv, bv, bv};
    #pragma unroll
    for (int kb = 0; kb < 4; ++kb) {
      const v16bf bf = *(const v16bf*)(C1p + ((kb * 8 + nb) * 32 + lane) * 16);
      acc = __builtin_amdgcn_wmma_f32_16x16x32_bf16(false, af[kb], false, bf,
                                                    (short)0, acc, false, false);
    }
    const int colb = nb * 16 + colLane;
    #pragma unroll
    for (int v = 0; v < 8; ++v)
      sHw[(rb + v) * ROWSTRIDE + colb] = (__bf16)fmaxf(acc[v], 0.0f);
  }

  // ====== layer 4 (classifier head as WMMA): cls = sigmoid(Hc@C2 + c2) ======
  // C2 is packed as a B-fragment column (only n==0 non-zero): 4 WMMAs replace
  // a 128-step redundant per-lane VALU dot. Column 0 lives in lanes 0 and 16.
  {
    load_afrags(sHw, af);
    v8f acc = {0.f, 0.f, 0.f, 0.f, 0.f, 0.f, 0.f, 0.f};
    #pragma unroll
    for (int kb = 0; kb < 4; ++kb) {
      const v16bf bf = *(const v16bf*)(C2p + kb * 512 + lane * 16);
      acc = __builtin_amdgcn_wmma_f32_16x16x32_bf16(false, af[kb], false, bf,
                                                    (short)0, acc, false, false);
    }
    if (colLane == 0) {            // lanes 0 (rows 0-7) and 16 (rows 8-15)
      #pragma unroll
      for (int v = 0; v < 8; ++v) {
        const int tr = rb + v;
        const int gr = rowBase + tr;
        if (gr < count) {
          const int item = sItems[wave][tr];
          const float cls = 1.0f / (1.0f + __expf(-(acc[v] + c2s)));
          out_score[item] = fminf(scores[item], cls);
          out_prob[item]  = best_p[s];
        }
      }
    }
  }
}

// ---------------------------------------------------------------------------
extern "C" void kernel_launch(void* const* d_in, const int* in_sizes, int n_in,
                              void* d_out, int out_size, void* d_ws, size_t ws_size,
                              hipStream_t stream) {
  const float* states   = (const float*)d_in[0];
  const float* scores   = (const float*)d_in[1];
  const int*   type_ids = (const int*)  d_in[2];
  const float* tm       = (const float*)d_in[3];
  const float* W1       = (const float*)d_in[4];
  const float* b1       = (const float*)d_in[5];
  const float* W2       = (const float*)d_in[6];
  const float* b2       = (const float*)d_in[7];
  const float* C1       = (const float*)d_in[8];
  const float* c1       = (const float*)d_in[9];
  const float* C2       = (const float*)d_in[10];
  const float* c2       = (const float*)d_in[11];

  const int N = in_sizes[2];               // 65536
  char* ws = (char*)d_ws;

  float* out_state = (float*)d_out;        // [N,128]
  float* out_score = out_state + (size_t)N * DIM;        // [N,1]
  float* out_prob  = out_state + (size_t)N * (DIM + 1);  // [N]

  prep_kernel<<<1, 256, 0, stream>>>(tm, W1, b1, W2, b2, C1, c1, C2, c2, ws);
  count_kernel<<<(N + 255) / 256, 256, 0, stream>>>(type_ids, ws, N);
  scan_kernel<<<1, 1, 0, stream>>>(ws);
  scatter_kernel<<<(N + 255) / 256, 256, 0, stream>>>(type_ids, ws, N);

  const int blocksPerType = (N + 63) / 64;  // 64 rows per block (4 waves x 16)
  expert_kernel<<<S_TYPES * blocksPerType, 128, 0, stream>>>(
      states, scores, ws, out_state, out_score, out_prob, N, blocksPerType);
}
